// SS2D_18391049962059
// MI455X (gfx1250) — matrix operand order repaired and checked by reading, not compiled, not run on hardware
//
#include <hip/hip_runtime.h>
#include <hip/hip_bf16.h>
#include <math.h>

#define Dd 96
#define Hh 96
#define Ww 96
#define LL 9216
#define KK 4
#define CPROJ 38
#define CP 48        // padded projection rows (3 WMMA M-tiles)
#define NN 16
#define RR 6
#define NCH 72       // scan chunks per (k,d)
#define LCH 128      // chunk length
#define APAD 104     // padded LDS row stride (bf16): 52 dwords, conflict-free
                     // across 16 lanes (gcd-driven: 52n mod 64 distinct for n=0..15)

typedef __attribute__((ext_vector_type(16))) __bf16 v16bf;
typedef __attribute__((ext_vector_type(8)))  float  v8f;
typedef __attribute__((ext_vector_type(4)))  unsigned int v4u;
typedef __attribute__((ext_vector_type(8)))  int  v8i_;
typedef __attribute__((ext_vector_type(4)))  int  v4i_;

__device__ __forceinline__ float softplus_f(float z) {
  return (z > 20.0f) ? z : log1pf(__expf(z));
}

// xs[k][d][l] = x[d*LL + dir_src(k,l)]
__device__ __forceinline__ int dir_src(int k, int l) {
  switch (k) {
    case 0: return l;
    case 1: return (l % Hh) * Ww + (l / Hh);
    case 2: return LL - 1 - l;
    default: { int lr = LL - 1 - l; return (lr % Hh) * Ww + (lr / Hh); }
  }
}

// ---------------------------------------------------------------------------
// Kernel 1: x_dbl[k][c][l] = sum_d W[k][c][d] * xs[k][d][l]  via bf16 WMMA.
// grid (LL/128, KK), block 256 (8 waves); wave w owns a 48x16 tile.
// k=0/2 tiles are rectangular in memory -> fetched by the Tensor Data Mover
// (tensor_load_to_lds, TENSORcnt); k=1/3 (transposed walks) gathered manually.
// ---------------------------------------------------------------------------
__global__ __launch_bounds__(256) void proj_kernel(
    const float* __restrict__ x, const float* __restrict__ wproj,
    float* __restrict__ xdbl) {
  __shared__ __bf16 Abf[CP][APAD];     // weights, zero padded rows 38..47
  __shared__ float  Bf32[Dd][128];     // raw xs tile (f32), TDM target
  __shared__ __bf16 Bt[128][APAD];     // transposed bf16 tile: Bt[lc][d]
  const int k    = blockIdx.y;
  const int lblk = blockIdx.x * 128;
  const int tid  = threadIdx.x;

  // ---- issue TDM load of the x tile for the row-major directions ----------
  if ((k == 0 || k == 2) && tid < 32) {          // wave 0 only
    const int col0 = (k == 0) ? lblk : (LL - 128 - lblk);  // k=2: reversed span
    unsigned long long ga = (unsigned long long)(const void*)(x + col0);
    unsigned int ldsa = (unsigned int)(unsigned long long)(void*)&Bf32[0][0];
    // D# group0: count=1 | lds_addr | global_addr[56:0] | type=2
    v4u g0 = { 1u, ldsa, (unsigned int)ga,
               (unsigned int)((ga >> 32) & 0x01FFFFFFu) | 0x80000000u };
    // D# group1: wg_mask=0, data_size=2(4B); tensor_dim0=9216; tensor_dim1=96;
    //            tile_dim0=128; tile_dim1=96; tensor_dim0_stride=9216
    v8i_ g1 = { 0x00020000,
                (int)(9216u << 16),       // tensor_dim0[15:0] in [31:16]
                (int)(96u << 16),         // dim0[31:16]=0 | tensor_dim1[15:0]
                (int)(128u << 16),        // dim1[31:16]=0 | tile_dim0
                96,                       // tile_dim1 | tile_dim2=0
                9216,                     // tensor_dim0_stride[31:0]
                0, 0 };                   // stride hi | tensor_dim1_stride=0
    v4i_ g2 = { 1, 1, 0, 0 };             // tensor_dim2=1, tensor_dim3=1 (benign)
    v4i_ g3 = { 0, 0x00010000, 0, 0 };    // tensor_dim4=1, tile_dim4=0
#if defined(__clang_major__) && __clang_major__ >= 23
    v8i_ z8 = { 0, 0, 0, 0, 0, 0, 0, 0 };
    __builtin_amdgcn_tensor_load_to_lds(g0, g1, g2, g3, z8, 0);
#else
    __builtin_amdgcn_tensor_load_to_lds(g0, g1, g2, g3, 0);
#endif
  }

  // ---- stage A (weights) while the DMA flies ------------------------------
  for (int i = tid; i < CP * Dd; i += 256) {
    int c = i / Dd, d = i % Dd;
    float v = (c < CPROJ) ? wproj[(size_t)(k * CPROJ + c) * Dd + d] : 0.0f;
    Abf[c][d] = (__bf16)v;
  }
  // ---- manual gather for the transposed directions ------------------------
  if (k == 1 || k == 3) {
    for (int i = tid; i < Dd * 128; i += 256) {
      int d = i >> 7, lc = i & 127;
      Bf32[d][lc] = x[(size_t)d * LL + dir_src(k, lblk + lc)];
    }
  }
  if ((k == 0 || k == 2) && tid < 32)
    __builtin_amdgcn_s_wait_tensorcnt(0);
  __syncthreads();

  // ---- f32 -> bf16 transpose into Bt[lc][d]; absorb k=2 reversal ----------
  for (int i = tid; i < 128 * Dd; i += 256) {
    int lc = i & 127, d = i >> 7;
    int rc = (k == 2) ? (127 - lc) : lc;
    Bt[lc][d] = (__bf16)Bf32[d][rc];
  }
  __syncthreads();

  const int wave = tid >> 5;
  const int lane = tid & 31;
  const int g    = lane >> 4;   // half-wave group
  const int mn   = lane & 15;   // M row (for A) / N col (for B,D)
  const int lw   = wave * 16;   // this wave's l-subtile within the block

  v8f acc0 = {}, acc1 = {}, acc2 = {};
  for (int kk = 0; kk < 3; ++kk) {   // K = 96 in steps of 32
    const int kb = kk * 32;
    v16bf b, a0, a1, a2;
#pragma unroll
    for (int j = 0; j < 16; ++j)
      b[j] = Bt[lw + mn][kb + g * 16 + j];           // B(kd, n): 16 contiguous
#pragma unroll
    for (int j = 0; j < 16; ++j) {
      int kd = kb + g * 16 + j;  // A(m, kd)
      a0[j] = Abf[mn][kd];
      a1[j] = Abf[16 + mn][kd];
      a2[j] = Abf[32 + mn][kd];
    }
    acc0 = __builtin_amdgcn_wmma_f32_16x16x32_bf16(false, a0, false, b, (short)0, acc0, false, false);
    acc1 = __builtin_amdgcn_wmma_f32_16x16x32_bf16(false, a1, false, b, (short)0, acc1, false, false);
    acc2 = __builtin_amdgcn_wmma_f32_16x16x32_bf16(false, a2, false, b, (short)0, acc2, false, false);
  }
  const int l = lblk + lw + mn;
#pragma unroll
  for (int r = 0; r < 8; ++r) {      // D(g*8+r, n)
    xdbl[((size_t)(k * CP +      (g << 3) + r)) * LL + l] = acc0[r];
    xdbl[((size_t)(k * CP + 16 + (g << 3) + r)) * LL + l] = acc1[r];
    xdbl[((size_t)(k * CP + 32 + (g << 3) + r)) * LL + l] = acc2[r];
  }
}

// ---------------------------------------------------------------------------
// Kernel 2: per-chunk local scan -> (prod dA, h_local). d fastest across lanes
// so dts/Bs loads are wave-uniform broadcasts.
// ---------------------------------------------------------------------------
__global__ __launch_bounds__(256) void scan_pass1(
    const float* __restrict__ x, const float* __restrict__ dtw,
    const float* __restrict__ dtb, const float* __restrict__ Alogs,
    const float* __restrict__ xdbl, float* __restrict__ chunkP,
    float* __restrict__ chunkH) {
  int gid = blockIdx.x * 256 + threadIdx.x;
  if (gid >= KK * NCH * Dd) return;
  int d = gid % Dd;
  int rest = gid / Dd;
  int chunk = rest % NCH;
  int k = rest / NCH;
  int kd = k * Dd + d;

  float A[NN], P[NN], h[NN];
#pragma unroll
  for (int nn = 0; nn < NN; ++nn) {
    A[nn] = -__expf(Alogs[kd * NN + nn]);
    P[nn] = 1.0f;
    h[nn] = 0.0f;
  }
  float w[RR];
#pragma unroll
  for (int r = 0; r < RR; ++r) w[r] = dtw[kd * RR + r];
  const float bias = dtb[kd];
  const float* dts  = xdbl + (size_t)(k * CP) * LL;          // rows 0..5
  const float* Bsr  = xdbl + (size_t)(k * CP + RR) * LL;     // rows 6..21
  const float* xrow = x + (size_t)d * LL;

  const int l0 = chunk * LCH;
  for (int li = 0; li < LCH; ++li) {
    int l = l0 + li;
    __builtin_prefetch(dts + l + 16, 0, 0);
    float dt = bias;
#pragma unroll
    for (int r = 0; r < RR; ++r) dt += w[r] * dts[(size_t)r * LL + l];
    float delta = softplus_f(dt);
    float xv = xrow[dir_src(k, l)];
    float dx = delta * xv;
#pragma unroll
    for (int nn = 0; nn < NN; ++nn) {
      float a  = __expf(delta * A[nn]);
      float bu = dx * Bsr[(size_t)nn * LL + l];
      P[nn] *= a;
      h[nn] = a * h[nn] + bu;
    }
  }
  size_t base = ((size_t)kd * NCH + chunk) * NN;
#pragma unroll
  for (int nn = 0; nn < NN; ++nn) {
    chunkP[base + nn] = P[nn];
    chunkH[base + nn] = h[nn];
  }
}

// ---------------------------------------------------------------------------
// Kernel 3: carry recurrence across the 72 chunks per (k,d,n)
// ---------------------------------------------------------------------------
__global__ __launch_bounds__(256) void scan_pass2(
    const float* __restrict__ chunkP, const float* __restrict__ chunkH,
    float* __restrict__ carry) {
  int gid = blockIdx.x * 256 + threadIdx.x;   // kd*NN + n
  if (gid >= KK * Dd * NN) return;
  int nn = gid % NN;
  int kd = gid / NN;
  float c = 0.0f;
  for (int ch = 0; ch < NCH; ++ch) {
    size_t idx = ((size_t)kd * NCH + ch) * NN + nn;
    carry[idx] = c;
    c = chunkP[idx] * c + chunkH[idx];
  }
}

// ---------------------------------------------------------------------------
// Kernel 4: replay with carry-in, contract with Cs, add skip: y stored [l][k*D+d]
// ---------------------------------------------------------------------------
__global__ __launch_bounds__(256) void scan_pass3(
    const float* __restrict__ x, const float* __restrict__ dtw,
    const float* __restrict__ dtb, const float* __restrict__ Alogs,
    const float* __restrict__ Dsv, const float* __restrict__ xdbl,
    const float* __restrict__ carry, float* __restrict__ yws) {
  int gid = blockIdx.x * 256 + threadIdx.x;
  if (gid >= KK * NCH * Dd) return;
  int d = gid % Dd;
  int rest = gid / Dd;
  int chunk = rest % NCH;
  int k = rest / NCH;
  int kd = k * Dd + d;

  float A[NN], h[NN];
  size_t cbase = ((size_t)kd * NCH + chunk) * NN;
#pragma unroll
  for (int nn = 0; nn < NN; ++nn) {
    A[nn] = -__expf(Alogs[kd * NN + nn]);
    h[nn] = carry[cbase + nn];
  }
  float w[RR];
#pragma unroll
  for (int r = 0; r < RR; ++r) w[r] = dtw[kd * RR + r];
  const float bias = dtb[kd];
  const float Dscale = Dsv[kd];
  const float* dts  = xdbl + (size_t)(k * CP) * LL;
  const float* Bsr  = xdbl + (size_t)(k * CP + RR) * LL;
  const float* Csr  = xdbl + (size_t)(k * CP + RR + NN) * LL;  // rows 22..37
  const float* xrow = x + (size_t)d * LL;

  const int l0 = chunk * LCH;
  for (int li = 0; li < LCH; ++li) {
    int l = l0 + li;
    __builtin_prefetch(Csr + l + 16, 0, 0);
    float dt = bias;
#pragma unroll
    for (int r = 0; r < RR; ++r) dt += w[r] * dts[(size_t)r * LL + l];
    float delta = softplus_f(dt);
    float xv = xrow[dir_src(k, l)];
    float dx = delta * xv;
    float yacc = xv * Dscale;
#pragma unroll
    for (int nn = 0; nn < NN; ++nn) {
      float a  = __expf(delta * A[nn]);
      float bu = dx * Bsr[(size_t)nn * LL + l];
      h[nn] = a * h[nn] + bu;
      yacc += h[nn] * Csr[(size_t)nn * LL + l];
    }
    yws[(size_t)l * (KK * Dd) + kd] = yacc;
  }
}

// ---------------------------------------------------------------------------
// Kernel 5: direction merge + LayerNorm over D, one wave per spatial position
// ---------------------------------------------------------------------------
__global__ __launch_bounds__(256) void combine_ln(
    const float* __restrict__ yws, const float* __restrict__ lnw,
    const float* __restrict__ lnb, float* __restrict__ out) {
  const int wave = threadIdx.x >> 5;
  const int lane = threadIdx.x & 31;
  const int p = blockIdx.x * 8 + wave;
  if (p >= LL) return;
  const int hh = p / Ww, ww = p % Ww;
  const int pt = ww * Hh + hh;            // transposed position for dirs 1,3
  float v[3];
#pragma unroll
  for (int i = 0; i < 3; ++i) {
    int d = lane + i * 32;
    v[i] = yws[(size_t)p * 384 + d]                         // y0[d][p]
         + yws[(size_t)(LL - 1 - p) * 384 + 192 + d]        // y2[d][L-1-p]
         + yws[(size_t)pt * 384 + 96 + d]                   // y1[d][pt]
         + yws[(size_t)(LL - 1 - pt) * 384 + 288 + d];      // y3[d][L-1-pt]
  }
  float s  = v[0] + v[1] + v[2];
  float sq = v[0] * v[0] + v[1] * v[1] + v[2] * v[2];
#pragma unroll
  for (int off = 16; off > 0; off >>= 1) {
    s  += __shfl_xor(s, off, 32);
    sq += __shfl_xor(sq, off, 32);
  }
  const float mu  = s * (1.0f / 96.0f);
  const float var = sq * (1.0f / 96.0f) - mu * mu;
  const float rs  = rsqrtf(var + 1e-5f);
#pragma unroll
  for (int i = 0; i < 3; ++i) {
    int d = lane + i * 32;
    out[(size_t)p * Dd + d] = (v[i] - mu) * rs * lnw[d] + lnb[d];
  }
}

extern "C" void kernel_launch(void* const* d_in, const int* in_sizes, int n_in,
                              void* d_out, int out_size, void* d_ws, size_t ws_size,
                              hipStream_t stream) {
  (void)in_sizes; (void)n_in; (void)out_size; (void)ws_size;
  const float* x   = (const float*)d_in[0];
  const float* wpr = (const float*)d_in[1];
  const float* dtw = (const float*)d_in[2];
  const float* dtb = (const float*)d_in[3];
  const float* Al  = (const float*)d_in[4];
  const float* Ds  = (const float*)d_in[5];
  const float* lnw = (const float*)d_in[6];
  const float* lnb = (const float*)d_in[7];

  float* ws     = (float*)d_ws;
  float* xdbl   = ws;                                      // K*48*L      = 1,769,472 f
  float* chunkP = xdbl   + (size_t)KK * CP * LL;           // K*D*72*16   =   442,368 f
  float* chunkH = chunkP + (size_t)KK * Dd * NCH * NN;
  float* carry  = chunkH + (size_t)KK * Dd * NCH * NN;
  float* yws    = carry  + (size_t)KK * Dd * NCH * NN;     // L*K*D       = 3,538,944 f
  // total ~25.3 MiB of workspace

  proj_kernel<<<dim3(LL / 128, KK), 256, 0, stream>>>(x, wpr, xdbl);
  const int t1 = KK * NCH * Dd;   // 27648
  scan_pass1<<<(t1 + 255) / 256, 256, 0, stream>>>(x, dtw, dtb, Al, xdbl, chunkP, chunkH);
  const int t2 = KK * Dd * NN;    // 6144
  scan_pass2<<<(t2 + 255) / 256, 256, 0, stream>>>(chunkP, chunkH, carry);
  scan_pass3<<<(t1 + 255) / 256, 256, 0, stream>>>(x, dtw, dtb, Al, Ds, xdbl, carry, yws);
  combine_ln<<<LL / 8, 256, 0, stream>>>(yws, lnw, lnb, (float*)d_out);
}
